// Model_42434276884499
// MI455X (gfx1250) — matrix operand order repaired and checked by reading, not compiled, not run on hardware
//
#include <hip/hip_runtime.h>
#include <hip/hip_bf16.h>

#define VOCAB  50000
#define EMB    256
#define HID    512
#define KTAGS  20
#define TSTART 18
#define TEND   19
#define BATCH  32
#define TLEN   256
#define NEGINF -10000.0f
#define NBLOCKS_LSTM 16

typedef _Float16 half8  __attribute__((ext_vector_type(8)));
typedef _Float16 half16 __attribute__((ext_vector_type(16)));
typedef float    float8 __attribute__((ext_vector_type(8)));

union H16U { half16 v; half8 h[2]; };

__device__ __forceinline__ half16 ld_a32(const _Float16* p0, const _Float16* p1) {
  H16U u;
  u.h[0] = *(const half8*)p0;
  u.h[1] = *(const half8*)p1;
  return u.v;
}

// Defeat LICM/PRE without losing global address-space provenance: make only
// the element OFFSET opaque (volatile asm can't be hoisted out of the t-loop),
// so the weight tiles are re-streamed from L2 each step as global_load_b128
// instead of being hoisted + spilled to scratch (round 2) or demoted to
// flat_load (round 3's opaque-pointer version).
__device__ __forceinline__ size_t opaque_off(size_t v) {
  asm volatile("" : "+v"(v));
  return v;
}

__device__ __forceinline__ float sigf(float x) { return 1.0f / (1.0f + __expf(-x)); }

// ---------------------------------------------------------------------------
// Prep: concatenated f16 weights Wcat[dir][4H][EMB+HID] = [Wih | Whh]
// ---------------------------------------------------------------------------
__global__ void prep_wcat(const float* __restrict__ Wf_ih, const float* __restrict__ Wf_hh,
                          const float* __restrict__ Wb_ih, const float* __restrict__ Wb_hh,
                          _Float16* __restrict__ wcat) {
  int idx = blockIdx.x * blockDim.x + threadIdx.x;
  const int per_dir = 4 * HID * (EMB + HID);
  if (idx >= 2 * per_dir) return;
  int dir = idx / per_dir;
  int rem = idx - dir * per_dir;
  int n  = rem / (EMB + HID);
  int kc = rem - n * (EMB + HID);
  const float* Wih = dir ? Wb_ih : Wf_ih;
  const float* Whh = dir ? Wb_hh : Wf_hh;
  float v = (kc < EMB) ? Wih[(size_t)n * EMB + kc] : Whh[(size_t)n * HID + (kc - EMB)];
  wcat[idx] = (_Float16)v;
}

// ---------------------------------------------------------------------------
// Prep: bias sums, padded f16 Wout, zero h state (both ping-pong slots),
// zero the grid-barrier counter.
// ---------------------------------------------------------------------------
__global__ void prep_misc(const float* __restrict__ bf_ih, const float* __restrict__ bf_hh,
                          const float* __restrict__ bb_ih, const float* __restrict__ bb_hh,
                          const float* __restrict__ Wout,
                          float* __restrict__ bsum, _Float16* __restrict__ woutp,
                          _Float16* __restrict__ hbuf, int* __restrict__ barrier_cnt) {
  int idx = blockIdx.x * blockDim.x + threadIdx.x;
  if (idx < 2 * 4 * HID) {
    int dir = idx / (4 * HID), n = idx % (4 * HID);
    bsum[idx] = dir ? (bb_ih[n] + bb_hh[n]) : (bf_ih[n] + bf_hh[n]);
    return;
  }
  idx -= 2 * 4 * HID;
  if (idx < 32 * 2 * HID) {
    int n = idx / (2 * HID), k = idx % (2 * HID);
    woutp[idx] = (n < KTAGS) ? (_Float16)Wout[(size_t)n * (2 * HID) + k] : (_Float16)0.0f;
    return;
  }
  idx -= 32 * 2 * HID;
  if (idx < 2 * 2 * BATCH * HID) { hbuf[idx] = (_Float16)0.0f; return; }
  idx -= 2 * 2 * BATCH * HID;
  if (idx == 0) { *barrier_cnt = 0; return; }
}

// ---------------------------------------------------------------------------
// Embedding gather -> f16 x (forward) and per-sample reversed x (backward)
// ---------------------------------------------------------------------------
__global__ void embed_kernel(const int* __restrict__ sentence, const int* __restrict__ lengths,
                             const float* __restrict__ emb,
                             _Float16* __restrict__ xf, _Float16* __restrict__ xb) {
  int idx = blockIdx.x * blockDim.x + threadIdx.x;
  if (idx >= BATCH * TLEN * EMB) return;
  int e  = idx % EMB;
  int bt = idx / EMB;
  int t  = bt % TLEN;
  int b  = bt / TLEN;
  int L  = lengths[b];
  int tok  = sentence[b * TLEN + t];
  xf[idx] = (_Float16)emb[(size_t)tok * EMB + e];
  int rt   = (t < L) ? (L - 1 - t) : t;
  int tokb = sentence[b * TLEN + rt];
  xb[idx] = (_Float16)emb[(size_t)tokb * EMB + e];
}

// ---------------------------------------------------------------------------
// Persistent BiLSTM: one launch for all T steps, both directions.
// 128 waves (16 blocks): wave -> (dir, mtile, ntile); each wave owns the
// 16x16 z-tiles of all 4 gates. Cell state c lives in 8 VGPRs per lane for
// the whole sequence. h ping-pongs through L2 with one device-scope barrier
// per timestep. k-loop is software-pipelined (double-buffered A/B registers);
// weight offsets are re-opaqued each step so the 96 B-tiles stream from L2.
// ---------------------------------------------------------------------------
__global__ __launch_bounds__(256)
void lstm_persistent(const _Float16* __restrict__ xf, const _Float16* __restrict__ xb,
                     const _Float16* __restrict__ wcat, const float* __restrict__ bsum,
                     _Float16* __restrict__ hbuf, _Float16* __restrict__ lstm_out,
                     const int* __restrict__ lengths, int* __restrict__ barrier_cnt) {
  const int lane  = threadIdx.x & 31;
  const int wv    = blockIdx.x * 8 + (threadIdx.x >> 5);   // 0..127
  const int dir   = wv >> 6;
  const int rem   = wv & 63;
  const int mtile = rem >> 5;        // batch row tile
  const int ntile = rem & 31;        // hidden col tile
  const int l15   = lane & 15;
  const int hi    = lane >> 4;

  const _Float16* x  = dir ? xb : xf;
  const float*    bs = bsum + dir * (4 * HID);

  const int arow = mtile * 16 + l15;   // A operand batch row
  const int ncol = ntile * 16 + l15;   // B/C operand hidden col
  const int c0 = hi ? 16 : 0;          // A lane K sub-offset
  const int kB = hi ? 16 : 0;          // B lane K sub-offset

  // hoisted per-lane constants
  const float bi = bs[0 * HID + ncol];
  const float bf = bs[1 * HID + ncol];
  const float bg = bs[2 * HID + ncol];
  const float bo = bs[3 * HID + ncol];
  int Lr[8];
#pragma unroll
  for (int r = 0; r < 8; ++r) Lr[r] = lengths[mtile * 16 + r + hi * 8];

  // element offsets (opaqued per timestep below)
  const size_t woff_base = (size_t)dir * (4 * HID) * (EMB + HID)
                         + (size_t)ncol * (EMB + HID) + kB;          // gate-0 row
  const size_t xoff_base = (size_t)arow * TLEN * EMB + c0;
  const size_t hoff_base = (size_t)arow * HID + c0;

  float c_reg[8] = {0.f, 0.f, 0.f, 0.f, 0.f, 0.f, 0.f, 0.f};

  for (int t = 0; t < TLEN; ++t) {
    const int pp = t & 1;
    const size_t hslot_prev = (size_t)(dir * 2 + pp)       * BATCH * HID;
    _Float16*    hnext = hbuf + (size_t)(dir * 2 + (pp ^ 1)) * BATCH * HID;

    // opaque per-iteration offsets: loads below cannot be hoisted across t,
    // but keep global address-space provenance (global_load, not flat_load)
    const _Float16* Wt   = wcat + opaque_off(woff_base);
    const _Float16* xcur = x    + opaque_off(xoff_base + (size_t)t * EMB);
    const _Float16* hrow = hbuf + opaque_off(hslot_prev + hoff_base);

    const _Float16* wrow[4];
#pragma unroll
    for (int g = 0; g < 4; ++g) wrow[g] = Wt + (size_t)g * HID * (EMB + HID);

    if (t + 1 < TLEN) __builtin_prefetch(x + xoff_base + (size_t)(t + 1) * EMB, 0, 1);

    float8 acc[4] = {};
    // --- software-pipelined GEMM over K = [x(256) | h(512)] ---
    half16 Acur = ld_a32(xcur, xcur + 8);
    half16 Bcur[4];
#pragma unroll
    for (int g = 0; g < 4; ++g) Bcur[g] = ld_a32(wrow[g], wrow[g] + 8);

#pragma unroll
    for (int kk = 0; kk < 24; ++kk) {
      half16 Anext = {};
      half16 Bnext[4] = {};
      if (kk < 23) {
        const int kb2 = (kk + 1) * 32;
        const _Float16* ar = (kk + 1 < 8) ? (xcur + kb2) : (hrow + (kb2 - EMB));
        Anext = ld_a32(ar, ar + 8);
#pragma unroll
        for (int g = 0; g < 4; ++g) {
          const _Float16* wp = wrow[g] + kb2;
          Bnext[g] = ld_a32(wp, wp + 8);
        }
      }
#pragma unroll
      for (int g = 0; g < 4; ++g)
        acc[g] = __builtin_amdgcn_wmma_f32_16x16x32_f16(false, Acur, false, Bcur[g],
                                                        (short)0, acc[g], false, false);
      Acur = Anext;
#pragma unroll
      for (int g = 0; g < 4; ++g) Bcur[g] = Bnext[g];
    }

    // --- gate math + cell update (c stays in registers) ---
#pragma unroll
    for (int r = 0; r < 8; ++r) {
      const int brow = mtile * 16 + r + hi * 8;
      const float zi = acc[0][r] + bi;
      const float zf = acc[1][r] + bf;
      const float zg = acc[2][r] + bg;
      const float zo = acc[3][r] + bo;
      const float cn = sigf(zf) * c_reg[r] + sigf(zi) * tanhf(zg);
      const float hn = sigf(zo) * tanhf(cn);
      c_reg[r] = cn;
      hnext[(size_t)brow * HID + ncol] = (_Float16)hn;
      const int L = Lr[r];
      const int wpos = dir ? ((t < L) ? (L - 1 - t) : t) : t;       // un-reverse bwd
      const _Float16 hm = (wpos < L) ? (_Float16)hn : (_Float16)0.0f; // pad mask
      lstm_out[((size_t)brow * TLEN + wpos) * (2 * HID) + dir * HID + ncol] = hm;
    }

    // --- device-scope grid barrier (16 co-resident blocks) ---
    __threadfence();            // release: every wave flushes its h stores
    __syncthreads();
    if (threadIdx.x == 0) {
      __hip_atomic_fetch_add(barrier_cnt, 1, __ATOMIC_RELEASE, __HIP_MEMORY_SCOPE_AGENT);
      const int target = NBLOCKS_LSTM * (t + 1);
      while (__hip_atomic_load(barrier_cnt, __ATOMIC_ACQUIRE, __HIP_MEMORY_SCOPE_AGENT) < target) {
        __builtin_amdgcn_s_sleep(1);
      }
      __threadfence();          // acquire: invalidate WGP cache
    }
    __syncthreads();
  }
}

// ---------------------------------------------------------------------------
// feats = lstm_out @ Wout^T + bout  : [8192,1024] x [1024,32(pad of 20)]
// 512 waves, one 16-row tile each, both N tiles per wave, pipelined k-loop.
// ---------------------------------------------------------------------------
__global__ __launch_bounds__(256)
void feats_kernel(const _Float16* __restrict__ lstm_out, const _Float16* __restrict__ woutp,
                  const float* __restrict__ bout, float* __restrict__ feats) {
  const int lane  = threadIdx.x & 31;
  const int wv    = blockIdx.x * 8 + (threadIdx.x >> 5);   // 0..511
  const int l15   = lane & 15;
  const int hi    = lane >> 4;
  const int mbase = wv * 16;
  const int c0 = hi ? 16 : 0;
  const int kB = hi ? 16 : 0;

  const _Float16* arow = lstm_out + (size_t)(mbase + l15) * (2 * HID) + c0;
  const _Float16* w0r  = woutp + (size_t)l15 * (2 * HID) + kB;
  const _Float16* w1r  = woutp + (size_t)(16 + l15) * (2 * HID) + kB;

  float8 acc0 = {}, acc1 = {};
  half16 Acur = ld_a32(arow, arow + 8);
  half16 B0c  = ld_a32(w0r, w0r + 8);
  half16 B1c  = ld_a32(w1r, w1r + 8);
#pragma unroll
  for (int kk = 0; kk < 32; ++kk) {
    half16 Anext = {}, B0n = {}, B1n = {};
    if (kk < 31) {
      const int kb2 = (kk + 1) * 32;
      Anext = ld_a32(arow + kb2, arow + kb2 + 8);
      B0n   = ld_a32(w0r + kb2, w0r + kb2 + 8);
      B1n   = ld_a32(w1r + kb2, w1r + kb2 + 8);
    }
    acc0 = __builtin_amdgcn_wmma_f32_16x16x32_f16(false, Acur, false, B0c, (short)0, acc0, false, false);
    acc1 = __builtin_amdgcn_wmma_f32_16x16x32_f16(false, Acur, false, B1c, (short)0, acc1, false, false);
    Acur = Anext; B0c = B0n; B1c = B1n;
  }
#pragma unroll
  for (int r = 0; r < 8; ++r) {
    const int m = mbase + r + hi * 8;
    feats[(size_t)m * KTAGS + l15] = acc0[r] + bout[l15];
    const int n1 = 16 + l15;
    if (n1 < KTAGS) feats[(size_t)m * KTAGS + n1] = acc1[r] + bout[n1];
  }
}

// ---------------------------------------------------------------------------
// Viterbi: one wave32 per batch sample, transitions in LDS, K=20 lanes active.
// Emits path_score [B] then best_path [B, T+1] (as floats) into d_out.
// ---------------------------------------------------------------------------
__global__ void viterbi_kernel(const float* __restrict__ feats, const float* __restrict__ trans,
                               const int* __restrict__ lengths, int* __restrict__ bptrs,
                               float* __restrict__ out) {
  const int b = blockIdx.x;
  const int k = threadIdx.x;   // 32 threads, single wave
  __shared__ float tr[KTAGS * KTAGS];
  __shared__ float fv[KTAGS];
  __shared__ float term[KTAGS];
  for (int i = k; i < KTAGS * KTAGS; i += 32) tr[i] = trans[i];
  if (k < KTAGS) fv[k] = (k == TSTART) ? 0.0f : NEGINF;
  __syncthreads();
  const int L = lengths[b];
  for (int t = 0; t < TLEN; ++t) {
    float fvn = 0.0f; int arg = 0;
    if (k < KTAGS) {
      float best = fv[0] + tr[k * KTAGS + 0];
      for (int p = 1; p < KTAGS; ++p) {
        float s = fv[p] + tr[k * KTAGS + p];
        if (s > best) { best = s; arg = p; }   // first-max tie-break like argmax
      }
      fvn = best + feats[((size_t)b * TLEN + t) * KTAGS + k];
      bptrs[((size_t)b * TLEN + t) * KTAGS + k] = arg;
    }
    __syncthreads();
    if (k < KTAGS) fv[k] = fvn;
    __syncthreads();
    if (t == L - 1 && k < KTAGS) term[k] = fvn + tr[TEND * KTAGS + k];
  }
  __syncthreads();
  if (k == 0) {
    float best = term[0]; int bt = 0;
    for (int p = 1; p < KTAGS; ++p) if (term[p] > best) { best = term[p]; bt = p; }
    out[b] = best;                                   // path_score [B,1]
    float* path = out + BATCH + (size_t)b * (TLEN + 1);
    path[TLEN] = (float)bt;
    int cur = bt;
    for (int t = TLEN - 1; t >= 0; --t) {
      const int src = t - (TLEN - L);                // right-aligned backpointers
      int nb;
      if (src >= 0) nb = (cur < KTAGS) ? bptrs[((size_t)b * TLEN + src) * KTAGS + cur] : KTAGS;
      else          nb = KTAGS;                      // sentinel = K
      path[t] = (float)nb;
      cur = nb;
    }
  }
}

// ---------------------------------------------------------------------------
extern "C" void kernel_launch(void* const* d_in, const int* in_sizes, int n_in,
                              void* d_out, int out_size, void* d_ws, size_t ws_size,
                              hipStream_t stream) {
  const int*   sentence = (const int*)d_in[0];
  const int*   lengths  = (const int*)d_in[1];
  const float* emb      = (const float*)d_in[2];
  const float* Wf_ih    = (const float*)d_in[3];
  const float* Wf_hh    = (const float*)d_in[4];
  const float* bf_ih    = (const float*)d_in[5];
  const float* bf_hh    = (const float*)d_in[6];
  const float* Wb_ih    = (const float*)d_in[7];
  const float* Wb_hh    = (const float*)d_in[8];
  const float* bb_ih    = (const float*)d_in[9];
  const float* bb_hh    = (const float*)d_in[10];
  const float* Wout     = (const float*)d_in[11];
  const float* bout     = (const float*)d_in[12];
  const float* trans    = (const float*)d_in[13];
  float* out = (float*)d_out;

  // workspace carve-up (256B-aligned offsets, ~33 MB total)
  char* ws = (char*)d_ws;
  _Float16* xf    = (_Float16*)(ws + 0);          //  4 MB  [B,T,EMB] f16
  _Float16* xb    = (_Float16*)(ws + 4194304);    //  4 MB  reversed x
  _Float16* wcat  = (_Float16*)(ws + 8388608);    //  6 MB  [2][4H][768] f16
  float*    bsum  = (float*)   (ws + 14680064);   // 16 KB  [2][4H]
  _Float16* woutp = (_Float16*)(ws + 14696448);   // 64 KB  [32][1024] f16
  _Float16* hbuf  = (_Float16*)(ws + 14761984);   // 128 KB [2][2][B][H] f16 (ping-pong)
  int*      bcnt  = (int*)     (ws + 14893056);   // 256 B  grid barrier counter
  _Float16* lout  = (_Float16*)(ws + 14893312);   // 16 MB  [B,T,2H] f16
  float*    feats = (float*)   (ws + 31670528);   // 640 KB [B,T,K] f32
  int*      bptr  = (int*)     (ws + 32325888);   // 640 KB [B,T,K] i32

  prep_wcat<<<(2 * 4 * HID * (EMB + HID) + 255) / 256, 256, 0, stream>>>(
      Wf_ih, Wf_hh, Wb_ih, Wb_hh, wcat);
  prep_misc<<<(2 * 4 * HID + 32 * 2 * HID + 2 * 2 * BATCH * HID + 1 + 255) / 256, 256, 0, stream>>>(
      bf_ih, bf_hh, bb_ih, bb_hh, Wout, bsum, woutp, hbuf, bcnt);
  embed_kernel<<<(BATCH * TLEN * EMB + 255) / 256, 256, 0, stream>>>(
      sentence, lengths, emb, xf, xb);

  lstm_persistent<<<NBLOCKS_LSTM, 256, 0, stream>>>(
      xf, xb, wcat, bsum, hbuf, lout, lengths, bcnt);

  feats_kernel<<<64, 256, 0, stream>>>(lout, woutp, bout, feats);
  viterbi_kernel<<<BATCH, 32, 0, stream>>>(feats, trans, lengths, bptr, out);
}
